// GroupedVectorAttention_63479616634986
// MI455X (gfx1250) — matrix operand order repaired and checked by reading, not compiled, not run on hardware
//
#include <hip/hip_runtime.h>
#include <hip/hip_bf16.h>

// GroupedVectorAttention for MI455X (gfx1250, wave32, WMMA bf16).
// N=40000 points, S=16 neighbors, C=96 channels, G=12 groups.
// One wave == one 16-row WMMA tile (16 neighbors of one point, or 16 points
// in the projection kernel).

typedef __attribute__((ext_vector_type(8)))  __bf16 v8bf;
typedef __attribute__((ext_vector_type(16))) __bf16 v16bf;
typedef __attribute__((ext_vector_type(8)))  float  v8f;

#define C_    96
#define S_    16
#define G_    12
#define LDA_  104      // padded bf16 row stride (16B-aligned: 104*2=208)
#define EPS_  1e-5f

__device__ __forceinline__ __bf16 f2bf(float f) {
  unsigned u = __builtin_bit_cast(unsigned, f);
  unsigned r = (u + 0x7FFFu + ((u >> 16) & 1u)) >> 16;   // round-to-nearest-even
  return __builtin_bit_cast(__bf16, (unsigned short)r);
}

// lane <-> lane^16 sum via ds_swizzle SWAPX16 (group-of-32: xor=0x10,and=0x1f)
__device__ __forceinline__ float xor16_sum(float x) {
  int o = __builtin_amdgcn_ds_swizzle(__builtin_bit_cast(int, x), 0x401F);
  return x + __builtin_bit_cast(float, o);
}

// A fragment (16x32 bf16, MxK) from row-major LDS tile, row stride LDA_.
// Per ISA layout: lane l holds row (l&15); element e maps to
// K = kc + e + (e&8) + ((l&16)?8:0)  -> two contiguous 16B chunks.
__device__ __forceinline__ v16bf load_a_frag(const __bf16* base, int kc, int lane) {
  int row  = lane & 15;
  int koff = (lane & 16) ? 8 : 0;
  const __bf16* p = base + row * LDA_ + kc + koff;
  v8bf lo = *(const v8bf*)p;          // K = kc+koff .. +7
  v8bf hi = *(const v8bf*)(p + 16);   // K = kc+koff+16 .. +23
  return __builtin_shufflevector(lo, hi, 0,1,2,3,4,5,6,7,8,9,10,11,12,13,14,15);
}

// B fragment (32x16 bf16, KxN) from LDS holding W^T: Wt[n][k], row stride LDA_.
// lane l holds column (l&15); element e -> K = kc + e + (e&8) + ((l&16)?8:0)
// -> two contiguous 16B chunks (same order as A).
__device__ __forceinline__ v16bf load_b_frag(const __bf16* base, int kc, int lane) {
  int n  = lane & 15;
  int kb = kc + ((lane & 16) ? 8 : 0);
  const __bf16* p = base + n * LDA_ + kb;
  v8bf lo = *(const v8bf*)p;
  v8bf hi = *(const v8bf*)(p + 16);
  return __builtin_shufflevector(lo, hi, 0,1,2,3,4,5,6,7,8,9,10,11,12,13,14,15);
}

#define WMMA_BF16(ACC, A, B) \
  (ACC) = __builtin_amdgcn_wmma_f32_16x16x32_bf16(false, (A), false, (B), (short)0, (ACC), false, false)

// ---------------------------------------------------------------------------
// Kernel 1: Y = maybe(relu(LN(X @ W + b, g, beta)))   X:[N,96] W:[96,96]
// 4 waves/block, each wave one 16-row tile.
// ---------------------------------------------------------------------------
__global__ __launch_bounds__(128) void gva_proj_kernel(
    const float* __restrict__ X, const float* __restrict__ W,
    const float* __restrict__ b, const float* __restrict__ g,
    const float* __restrict__ beta, float* __restrict__ Y, int N, int doLN)
{
  __shared__ __align__(16) __bf16 sWt[C_][LDA_];          // W^T bf16
  __shared__ float sB[C_], sG[C_], sBeta[C_];
  __shared__ __align__(16) __bf16 sX[4][S_][LDA_];        // per-wave A tiles
  __shared__ float sO[4][S_][C_];                         // per-wave D tiles

  const int tid  = threadIdx.x;
  const int wave = tid >> 5;
  const int lane = tid & 31;

  for (int i = tid; i < C_ * C_; i += 128) {
    int k = i / C_, n = i % C_;
    sWt[n][k] = f2bf(W[i]);
  }
  if (tid < C_) {
    sB[tid] = b[tid];
    if (doLN) { sG[tid] = g[tid]; sBeta[tid] = beta[tid]; }
  }

  int row0 = (blockIdx.x * 4 + wave) * 16;
  if (row0 > N - 16) row0 = N - 16;   // clamp (N%16==0); duplicate rows write same values

  { // stage X tile as bf16
    int r  = lane & 15;
    int c0 = (lane >> 4) * 48;
    const float* src = X + (size_t)(row0 + r) * C_ + c0;
    for (int j = 0; j < 48; ++j) sX[wave][r][c0 + j] = f2bf(src[j]);
  }
  __syncthreads();

  v8f acc[6];
  for (int t = 0; t < 6; ++t)
    for (int v = 0; v < 8; ++v) acc[t][v] = 0.f;

  for (int kc = 0; kc < C_; kc += 32) {
    v16bf a = load_a_frag(&sX[wave][0][0], kc, lane);
    for (int t = 0; t < 6; ++t) {
      v16bf bb = load_b_frag(&sWt[t * 16][0], kc, lane);
      WMMA_BF16(acc[t], a, bb);
    }
  }

  { // bias add, spill D tile to LDS for row-wise LN
    int cl = lane & 15;
    int m0 = (lane & 16) ? 8 : 0;
    for (int t = 0; t < 6; ++t) {
      int col = t * 16 + cl;
      float bias = sB[col];
      for (int v = 0; v < 8; ++v) sO[wave][m0 + v][col] = acc[t][v] + bias;
    }
  }
  __syncthreads();

  { // LN(+relu) per row, write out (2 lanes per row, 48 cols each)
    int r  = lane & 15;
    int c0 = (lane >> 4) * 48;
    float* dst = Y + (size_t)(row0 + r) * C_;
    if (doLN) {
      float s1 = 0.f, s2 = 0.f;
      for (int j = 0; j < 48; ++j) {
        float x = sO[wave][r][c0 + j];
        s1 += x; s2 += x * x;
      }
      s1 = xor16_sum(s1); s2 = xor16_sum(s2);
      float mu   = s1 * (1.f / C_);
      float var  = s2 * (1.f / C_) - mu * mu;
      float rstd = rsqrtf(var + EPS_);
      for (int j = 0; j < 48; ++j) {
        int c = c0 + j;
        float y = (sO[wave][r][c] - mu) * rstd * sG[c] + sBeta[c];
        dst[c] = y > 0.f ? y : 0.f;
      }
    } else {
      for (int j = 0; j < 48; ++j) { int c = c0 + j; dst[c] = sO[wave][r][c]; }
    }
  }
}

// ---------------------------------------------------------------------------
// Kernel 2: fused neighbor attention, one wave per point (M = 16 neighbors).
// ---------------------------------------------------------------------------
__global__ __launch_bounds__(128) void gva_attn_kernel(
    const float* __restrict__ xyz, const int* __restrict__ refidx,
    const float* __restrict__ query, const float* __restrict__ keyf,
    const float* __restrict__ val,
    const float* __restrict__ Wp1, const float* __restrict__ bp1,
    const float* __restrict__ gp,  const float* __restrict__ betap,
    const float* __restrict__ Wp2, const float* __restrict__ bp2,
    const float* __restrict__ Ww1, const float* __restrict__ bw1,
    const float* __restrict__ gw,  const float* __restrict__ betaw,
    const float* __restrict__ Ww2, const float* __restrict__ bw2,
    float* __restrict__ out, int N)
{
  __shared__ __align__(16) __bf16 sWp2t[C_][LDA_];   // Wp2^T bf16
  __shared__ __align__(16) __bf16 sWw1t[16][LDA_];   // Ww1^T bf16, cols 12..15 zero
  __shared__ float sWp1[3][C_];
  __shared__ float sbp1[C_], sbp2[C_], sgp[C_], sbetap[C_];
  __shared__ float sWw2[G_][G_];
  __shared__ float sbw1[G_], sgw[G_], sbetaw[G_], sbw2[G_];
  __shared__ __align__(16) __bf16 sA[4][S_][LDA_];   // h1 tile, then rel tile
  __shared__ float sT[4][S_][16];                    // rel@Ww1+bw1
  __shared__ float sW2[4][S_][16];                   // softmax weights
  __shared__ int   sIdx[4][S_];
  __shared__ float sMask[4][S_];

  const int tid  = threadIdx.x;
  const int wave = tid >> 5;
  const int lane = tid & 31;
  const int cl   = lane & 15;
  const int m0   = (lane & 16) ? 8 : 0;

  // ---- stage shared weights (bf16-transposed for the B operand) ----
  for (int i = tid; i < C_ * C_; i += 128) {
    int k = i / C_, n = i % C_;
    sWp2t[n][k] = f2bf(Wp2[i]);
  }
  for (int i = tid; i < 16 * C_; i += 128) {
    int n = i / C_, k = i % C_;
    sWw1t[n][k] = f2bf(n < G_ ? Ww1[k * G_ + n] : 0.f);
  }
  for (int i = tid; i < 3 * C_; i += 128) sWp1[i / C_][i % C_] = Wp1[i];
  if (tid < C_) { sbp1[tid] = bp1[tid]; sbp2[tid] = bp2[tid];
                  sgp[tid] = gp[tid];  sbetap[tid] = betap[tid]; }
  for (int i = tid; i < G_ * G_; i += 128) sWw2[i / G_][i % G_] = Ww2[i];
  if (tid < G_) { sbw1[tid] = bw1[tid]; sgw[tid] = gw[tid];
                  sbetaw[tid] = betaw[tid]; sbw2[tid] = bw2[tid]; }

  int n = blockIdx.x * 4 + wave;
  if (n > N - 1) n = N - 1;          // N%4==0 in practice; keeps barriers uniform

  if (lane < S_) {
    int iv = refidx[(size_t)n * S_ + lane];
    sIdx[wave][lane]  = iv < 0 ? 0 : iv;
    sMask[wave][lane] = iv < 0 ? 0.f : 1.f;
  }
  __syncthreads();

  const float xn0 = xyz[n * 3 + 0], xn1 = xyz[n * 3 + 1], xn2 = xyz[n * 3 + 2];

  // ---- h1 = relu(LN(pos @ Wp1 + bp1)) -> bf16 tile in sA ----
  {
    int s  = lane & 15;
    int c0 = (lane >> 4) * 48;
    int is = sIdx[wave][s];
    float mk = sMask[wave][s];
    float px = (xyz[is * 3 + 0] - xn0) * mk;
    float py = (xyz[is * 3 + 1] - xn1) * mk;
    float pz = (xyz[is * 3 + 2] - xn2) * mk;
    float s1 = 0.f, s2 = 0.f;
    for (int j = 0; j < 48; ++j) {
      int c = c0 + j;
      float h = sbp1[c] + px * sWp1[0][c] + py * sWp1[1][c] + pz * sWp1[2][c];
      s1 += h; s2 += h * h;
    }
    s1 = xor16_sum(s1); s2 = xor16_sum(s2);
    float mu   = s1 * (1.f / C_);
    float rstd = rsqrtf(s2 * (1.f / C_) - mu * mu + EPS_);
    for (int j = 0; j < 48; ++j) {
      int c = c0 + j;
      float h = sbp1[c] + px * sWp1[0][c] + py * sWp1[1][c] + pz * sWp1[2][c];
      float y = (h - mu) * rstd * sgp[c] + sbetap[c];
      sA[wave][s][c] = f2bf(y > 0.f ? y : 0.f);
    }
  }
  __syncthreads();

  // ---- peb = h1 @ Wp2 + bp2  (WMMA, 3 k-chunks x 6 n-tiles) ----
  v8f acc[6];
  for (int t = 0; t < 6; ++t)
    for (int v = 0; v < 8; ++v) acc[t][v] = 0.f;
  for (int kc = 0; kc < C_; kc += 32) {
    v16bf a = load_a_frag(&sA[wave][0][0], kc, lane);
    for (int t = 0; t < 6; ++t) {
      v16bf bb = load_b_frag(&sWp2t[t * 16][0], kc, lane);
      WMMA_BF16(acc[t], a, bb);
    }
  }
  for (int t = 0; t < 6; ++t) {
    float bias = sbp2[t * 16 + cl];
    for (int v = 0; v < 8; ++v) acc[t][v] += bias;
  }
  __syncthreads();

  // ---- rel = keyg - query + peb -> sA(bf16); valg = valgather + peb -> acc ----
  for (int t = 0; t < 6; ++t) {
    int col = t * 16 + cl;
    float qv = query[(size_t)n * C_ + col];
    for (int v = 0; v < 8; ++v) {
      int m  = m0 + v;
      int im = sIdx[wave][m];
      float mk  = sMask[wave][m];
      float peb = acc[t][v];
      float rel = keyf[(size_t)im * C_ + col] * mk - qv + peb;   // L2-resident gather
      sA[wave][m][col] = f2bf(rel);
      acc[t][v] = val[(size_t)im * C_ + col] * mk + peb;         // valg stays in regs
    }
  }
  __syncthreads();

  // ---- t1 = rel @ Ww1 + bw1  (WMMA, 12 cols padded to 16) ----
  {
    v8f acc2;
    for (int v = 0; v < 8; ++v) acc2[v] = 0.f;
    for (int kc = 0; kc < C_; kc += 32) {
      v16bf a  = load_a_frag(&sA[wave][0][0], kc, lane);
      v16bf bb = load_b_frag(&sWw1t[0][0], kc, lane);
      WMMA_BF16(acc2, a, bb);
    }
    for (int v = 0; v < 8; ++v)
      sT[wave][m0 + v][cl] = (cl < G_) ? acc2[v] + sbw1[cl] : 0.f;
  }
  __syncthreads();

  // ---- per-row LN over G=12 + relu, then @Ww2 + bw2 ----
  if (lane < S_) {
    int s = lane;
    float x[G_]; float s1 = 0.f, s2 = 0.f;
    for (int j = 0; j < G_; ++j) { x[j] = sT[wave][s][j]; s1 += x[j]; s2 += x[j] * x[j]; }
    float mu   = s1 * (1.f / G_);
    float rstd = rsqrtf(s2 * (1.f / G_) - mu * mu + EPS_);
    float xr[G_];
    for (int j = 0; j < G_; ++j) {
      float y = (x[j] - mu) * rstd * sgw[j] + sbetaw[j];
      xr[j] = y > 0.f ? y : 0.f;
    }
    for (int gg = 0; gg < G_; ++gg) {
      float h = sbw2[gg];
      for (int j = 0; j < G_; ++j) h += xr[j] * sWw2[j][gg];
      sW2[wave][s][gg] = h;
    }
  }
  __syncthreads();

  // ---- softmax over the 16 neighbors per group, * mask ----
  if (lane < G_) {
    int gg = lane;
    float mx = -1e30f;
    for (int s = 0; s < S_; ++s) mx = fmaxf(mx, sW2[wave][s][gg]);
    float e[S_]; float sum = 0.f;
    for (int s = 0; s < S_; ++s) { e[s] = expf(sW2[wave][s][gg] - mx); sum += e[s]; }
    float inv = 1.f / sum;
    for (int s = 0; s < S_; ++s) sW2[wave][s][gg] = e[s] * inv * sMask[wave][s];
  }
  __syncthreads();

  // ---- feat[col] = sum_s valg[s][col] * w[s][col/8]; combine lane-halves ----
  for (int t = 0; t < 6; ++t) {
    int col = t * 16 + cl;
    int gg  = col >> 3;          // C/G = 8
    float p = 0.f;
    for (int v = 0; v < 8; ++v) p += acc[t][v] * sW2[wave][m0 + v][gg];
    p = xor16_sum(p);
    if (!(lane & 16)) out[(size_t)n * C_ + col] = p;
  }
}

// ---------------------------------------------------------------------------
extern "C" void kernel_launch(void* const* d_in, const int* in_sizes, int n_in,
                              void* d_out, int out_size, void* d_ws, size_t ws_size,
                              hipStream_t stream) {
  const float* q    = (const float*)d_in[0];
  const float* k    = (const float*)d_in[1];
  const float* v    = (const float*)d_in[2];
  const float* xyz  = (const float*)d_in[3];
  const int*   ridx = (const int*)  d_in[4];
  const float* Wq   = (const float*)d_in[5];
  const float* bq   = (const float*)d_in[6];
  const float* gq   = (const float*)d_in[7];
  const float* betaq= (const float*)d_in[8];
  const float* Wk   = (const float*)d_in[9];
  const float* bk   = (const float*)d_in[10];
  const float* gk   = (const float*)d_in[11];
  const float* betak= (const float*)d_in[12];
  const float* Wv   = (const float*)d_in[13];
  const float* bv   = (const float*)d_in[14];
  const float* Wp1  = (const float*)d_in[15];
  const float* bp1  = (const float*)d_in[16];
  const float* gp   = (const float*)d_in[17];
  const float* betap= (const float*)d_in[18];
  const float* Wp2  = (const float*)d_in[19];
  const float* bp2  = (const float*)d_in[20];
  const float* Ww1  = (const float*)d_in[21];
  const float* bw1  = (const float*)d_in[22];
  const float* gw   = (const float*)d_in[23];
  const float* betaw= (const float*)d_in[24];
  const float* Ww2  = (const float*)d_in[25];
  const float* bw2  = (const float*)d_in[26];

  const int N = in_sizes[0] / C_;

  float* ws    = (float*)d_ws;
  float* qProj = ws;
  float* kProj = ws + (size_t)N * C_;
  float* vProj = ws + (size_t)2 * N * C_;

  const int projBlocks = ((N / 16) + 3) / 4;   // 4 tiles (waves) per block
  gva_proj_kernel<<<projBlocks, 128, 0, stream>>>(q, Wq, bq, gq, betaq, qProj, N, 1);
  gva_proj_kernel<<<projBlocks, 128, 0, stream>>>(k, Wk, bk, gk, betak, kProj, N, 1);
  gva_proj_kernel<<<projBlocks, 128, 0, stream>>>(v, Wv, bv, gq, betaq, vProj, N, 0);

  const int attnBlocks = (N + 3) / 4;          // 1 wave per point
  gva_attn_kernel<<<attnBlocks, 128, 0, stream>>>(
      xyz, ridx, qProj, kProj, vProj,
      Wp1, bp1, gp, betap, Wp2, bp2,
      Ww1, bw1, gw, betaw, Ww2, bw2,
      (float*)d_out, N);
}
